// SplatterModel_17454747091703
// MI455X (gfx1250) — matrix operand order repaired and checked by reading, not compiled, not run on hardware
//
#include <hip/hip_runtime.h>
#include <hip/hip_bf16.h>

typedef __attribute__((ext_vector_type(16))) _Float16 v16h;
typedef __attribute__((ext_vector_type(8)))  _Float16 v8h;
typedef __attribute__((ext_vector_type(8)))  float    v8f;
typedef __attribute__((ext_vector_type(4)))  float    v4f;
typedef __attribute__((ext_vector_type(2)))  float    v2f;

#define NSPLAT   1000
#define NPAD     1024
#define NCHUNK   (NPAD / 32)
#define KDIM     256
#define EPSV     1e-6f
#define LOG2E    1.4426950408889634f
// per-splat record: 8 floats = {a', b2', c', zmax', cx, cy, [col0,col1|col2,0 as 4xf16]}
#define REC      8
// ws layout: [0, 8192) floats = param records (32KB)
//            [8192, 16384) floats = packed A-matrix colors, 16384 f16 (32KB)
#define WS_A_OFF (NPAD * REC)

// ---------------------------------------------------------------------------
// Kernel 1: parameter transforms + analytic per-splat zmax (in log2 units).
// weight = exp2(z' - zmax'); the 2*pi*sqrt(det) normalizer cancels in kn.
// z concave in y => per grid row, max at floor/ceil of y* = -b2'x/(2c').
// ---------------------------------------------------------------------------
__global__ __launch_bounds__(256) void splat_prep(
    const float* __restrict__ rho_raw, const float* __restrict__ sigma_x_raw,
    const float* __restrict__ sigma_y_raw, const float* __restrict__ coords_raw,
    const float* __restrict__ alpha_raw, const float* __restrict__ colors_raw,
    float* __restrict__ ws) {
  const int i = blockIdx.x * 256 + threadIdx.x;
  if (i >= NPAD) return;

  float a = 0.f, b2 = 0.f, cc = 0.f, zm = 0.f, cx = 0.f, cy = 0.f;
  _Float16 c0 = (_Float16)0.f, c1 = (_Float16)0.f, c2 = (_Float16)0.f;

  if (i < NSPLAT) {
    const float alpha = 1.f / (1.f + __expf(-alpha_raw[i]));
    c0 = (_Float16)tanhf(colors_raw[3 * i + 0] * alpha);
    c1 = (_Float16)tanhf(colors_raw[3 * i + 1] * alpha);
    c2 = (_Float16)tanhf(colors_raw[3 * i + 2] * alpha);

    const float rho = tanhf(rho_raw[i]);
    const float sx  = -tanhf(sigma_x_raw[i]);
    const float sy  = -tanhf(sigma_y_raw[i]);
    cx = tanhf(coords_raw[2 * i + 0]);
    cy = tanhf(coords_raw[2 * i + 1]);

    const float e00 = sx * sx + EPSV;
    const float e01 = rho * sx * sy + EPSV;
    const float e11 = sy * sy + EPSV;
    const float det = e00 * e11 - e01 * e01;
    const float rdet = 1.f / det;
    a  = -0.5f * LOG2E * (e11 * rdet);
    b2 =          LOG2E * (e01 * rdet);
    cc = -0.5f * LOG2E * (e00 * rdet);

    const float stepg = 2.f / 255.f;
    const float denom = 2.f * cc;
    float best = -3.0e38f;
    for (int hh = 0; hh < KDIM; ++hh) {
      const float x = fmaf((float)hh, stepg, -1.f) + cx;
      const float ystar = (denom != 0.f) ? (-b2 * x / denom) : 0.f;
      const float jf = (ystar - cy + 1.f) * (255.f / 2.f);
      int j0 = (int)floorf(jf);
      #pragma unroll
      for (int dj = 0; dj <= 1; ++dj) {
        int j = min(255, max(0, j0 + dj));
        const float y = fmaf((float)j, stepg, -1.f) + cy;
        const float z = fmaf(x, fmaf(a, x, b2 * y), cc * y * y);
        best = fmaxf(best, z);
      }
    }
    zm = best;
  }

  float* p = ws + i * REC;
  p[0] = a; p[1] = b2; p[2] = cc; p[3] = zm; p[4] = cx; p[5] = cy;
  _Float16* hp = (_Float16*)(p + 6);
  hp[0] = c0; hp[1] = c1; hp[2] = c2; hp[3] = (_Float16)0.f;
}

// ---------------------------------------------------------------------------
// Kernel 2: pre-pack the color A-matrix operand in exact WMMA register layout.
// A (16x32 f16, M=channel, K=splat): lane L holds M=L%16; VGPR j holds
// K = 2(j&3) + 8*(L>=16) + 16*(j>>2), +1. Rows 3..15 are zero.
// Stored as [chunk][g=0,1][lane] rows of v8h so the main kernel fetches one
// chunk's operand with two unconditional ds_load_b128 (no exec juggling).
// ---------------------------------------------------------------------------
__global__ __launch_bounds__(256) void splat_packA(const float* __restrict__ ws,
                                                   _Float16* __restrict__ wsA) {
  const int i = blockIdx.x * 256 + threadIdx.x;   // 0..1023 -> (chunk, lane)
  const int chunk = i >> 5;
  const int lane  = i & 31;
  const int half  = lane >> 4;
  const int l15   = lane & 15;
  const _Float16* wsH = (const _Float16*)ws;      // record = 16 halves; colors at +12

  _Float16 vals[16];
  #pragma unroll
  for (int e = 0; e < 16; ++e) {
    const int j = e >> 1;
    const int K = ((j & 3) << 1) + (half << 3) + ((j >> 2) << 4) + (e & 1);
    const int s = (chunk << 5) + K;
    vals[e] = (l15 < 3) ? wsH[(s << 4) + 12 + l15] : (_Float16)0.f;
  }
  #pragma unroll
  for (int g = 0; g < 2; ++g) {
    v8h row;
    #pragma unroll
    for (int e = 0; e < 8; ++e) row[e] = vals[g * 8 + e];
    ((v8h*)wsA)[(chunk << 6) + (g << 5) + lane] = row;
  }
}

// weight for splat s at pixel (xh, yw), from the LDS splat table
__device__ __forceinline__ _Float16 splat_wgt(const float* __restrict__ sp, int s,
                                              float xh, float yw) {
  const v4f q  = *(const v4f*)(sp + (s << 3));        // a, b2, c, zmax
  const v2f ct = *(const v2f*)(sp + (s << 3) + 4);    // cx, cy
  const float x = xh + ct.x;
  const float y = yw + ct.y;
  const float z = fmaf(x, fmaf(q.x, x, q.y * y), q.z * y * y);
  return (_Float16)__builtin_amdgcn_exp2f(z - q.w);
}

// ---------------------------------------------------------------------------
// Kernel 3: fused weight-eval + WMMA color contraction.
// Wave owns 16 contiguous pixels of one row. Per 32-splat chunk:
//   A (16x32 f16)  = colors (pre-packed, 2x ds_load_b128)
//   B (32x16 f16)  = weights exp2(z'-zmax'), computed in-register
//   C/D (16x16 f32) accumulates out[channel][pixel] across 32 chunks.
// ---------------------------------------------------------------------------
__global__ __launch_bounds__(256) void splat_main(const float* __restrict__ ws,
                                                  float* __restrict__ out) {
  __shared__ float    sp[NPAD * REC];   // 32 KB param table
  __shared__ _Float16 sA[NPAD * 16];    // 32 KB packed A colors
  const int tid = threadIdx.x;

  {
    const v4f* g0 = (const v4f*)ws;
    v4f* s0 = (v4f*)sp;
    for (int i = tid; i < NPAD * REC / 4; i += 256) s0[i] = g0[i];
    const v4f* g1 = (const v4f*)(ws + WS_A_OFF);
    v4f* s1 = (v4f*)sA;
    for (int i = tid; i < NPAD * 16 / 8; i += 256) s1[i] = g1[i];
  }
  __syncthreads();

  const int lane = tid & 31;
  const int wid  = tid >> 5;
  const int half = lane >> 4;
  const int l15  = lane & 15;

  const int h = blockIdx.x >> 1;
  const int w = ((blockIdx.x & 1) << 7) + wid * 16 + l15;

  const float stepg = 2.f / 255.f;
  const float xh = fmaf((float)h, stepg, -1.f);
  const float yw = fmaf((float)w, stepg, -1.f);

  const v8h* sA8 = (const v8h*)sA;
  v8f acc = {};

  for (int chunk = 0; chunk < NCHUNK; ++chunk) {
    const int base = chunk << 5;

    // A: two unconditional b128 LDS loads in operand layout
    const v8h a0 = sA8[(chunk << 6) + lane];
    const v8h a1 = sA8[(chunk << 6) + 32 + lane];
    const v16h amat = __builtin_shufflevector(a0, a1,
        0, 1, 2, 3, 4, 5, 6, 7, 8, 9, 10, 11, 12, 13, 14, 15);

    // B layout (16-bit B 32x16): VGPR j holds rows K = 2(j&3) + 8*half + 16*(j>>2), +1 ; N = l15
    v16h bmat;
    #pragma unroll
    for (int j = 0; j < 8; ++j) {
      const int s0 = base + ((j & 3) << 1) + (half << 3) + ((j >> 2) << 4);
      bmat[2 * j]     = splat_wgt(sp, s0,     xh, yw);
      bmat[2 * j + 1] = splat_wgt(sp, s0 + 1, xh, yw);
    }

    acc = __builtin_amdgcn_wmma_f32_16x16x32_f16(
        /*neg_a=*/false, amat, /*neg_b=*/false, bmat,
        /*c_mod=*/(short)0, acc, /*reuse_a=*/false, /*reuse_b=*/false);
  }

  // D layout: lanes 0-15, VGPR r -> M=r, N=lane. Channels 0..2 live in acc[0..2].
  if (lane < 16) {
    float* o = out + (h * 256 + w) * 3;
    o[0] = acc[0];
    o[1] = acc[1];
    o[2] = acc[2];
  }
}

extern "C" void kernel_launch(void* const* d_in, const int* in_sizes, int n_in,
                              void* d_out, int out_size, void* d_ws, size_t ws_size,
                              hipStream_t stream) {
  (void)in_sizes; (void)n_in; (void)out_size; (void)ws_size;
  const float* rho_raw     = (const float*)d_in[0];
  const float* sigma_x_raw = (const float*)d_in[1];
  const float* sigma_y_raw = (const float*)d_in[2];
  const float* coords_raw  = (const float*)d_in[3];
  const float* alpha_raw   = (const float*)d_in[4];
  const float* colors_raw  = (const float*)d_in[5];
  float* out = (float*)d_out;
  float* ws  = (float*)d_ws;   // 64 KB used

  splat_prep<<<NPAD / 256, 256, 0, stream>>>(rho_raw, sigma_x_raw, sigma_y_raw,
                                             coords_raw, alpha_raw, colors_raw, ws);
  splat_packA<<<NPAD / 256, 256, 0, stream>>>(ws, (_Float16*)(ws + WS_A_OFF));
  splat_main<<<512, 256, 0, stream>>>(ws, out);
}
